// GNN_66340064854629
// MI455X (gfx1250) — matrix operand order repaired and checked by reading, not compiled, not run on hardware
//
#include <hip/hip_runtime.h>

// ---------------------------------------------------------------------------
// GIN encoder forward for MI455X (gfx1250, wave32, WMMA).
// fp32 everywhere except GEMM inputs (bf16 -> v_wmma_f32_16x16x32_bf16, f32
// accumulate). Working set ~80MB < 192MB L2 -> L2-bandwidth bound.
// Weights are packed ONCE per launch into fragment-major bf16 in global ws;
// GEMM waves read B fragments straight from L2/WGP$ (2x global_load_b128),
// A tiles are staged in LDS in natural K order, which keeps both the global
// reads (f32) and the LDS fragment loads (b128) fully vectorized.
// ---------------------------------------------------------------------------

#define GN 50000
#define GE 500000
#define GD 128
#define GL 4
#define GG 64

typedef __bf16 bf16_t;
typedef __attribute__((ext_vector_type(16))) __bf16 v16bf;
typedef __attribute__((ext_vector_type(8)))  __bf16 v8bf;
typedef __attribute__((ext_vector_type(8)))  float  v8f;

union Frag32B { v16bf v; v8bf h[2]; };

// ---------------------------------------------------------------------------
// Zero the small accumulators we rely on each launch (deterministic start).
__global__ __launch_bounds__(256)
void zero_misc(int* __restrict__ indeg, float* __restrict__ gsum,
               int* __restrict__ gcnt, float* __restrict__ stats) {
  int i = blockIdx.x * blockDim.x + threadIdx.x;
  if (i < GN)     indeg[i] = 0;
  if (i < GG*GD)  gsum[i]  = 0.f;
  if (i < GG)     gcnt[i]  = 0;
  if (i < 2*GD)   stats[i] = 0.f;
}

// ---------------------------------------------------------------------------
// Pack one 128x128 f32 weight matrix per block into fragment-major bf16:
//   dst[((t*4+kk)*32 + lane)*16 + j] = W[(kk*32 + (lane>=16)*16 + j)*128 + t*16 + lane%16]
// so a WMMA B fragment is a contiguous 32B per lane (lane half h holds k = 16*h + j of each 32-k step, as A).
__global__ __launch_bounds__(128)
void pack_w(const float* __restrict__ W, bf16_t* __restrict__ Wp) {
  const float* w = W + (size_t)blockIdx.x * GD * GD;
  bf16_t*      o = Wp + (size_t)blockIdx.x * GD * GD;
  for (int idx = threadIdx.x; idx < GD * GD; idx += 128) {
    int k = idx >> 7, n = idx & 127;
    int t = n >> 4, c = n & 15;
    int kk = k >> 5, rem = k & 31, h2 = rem >> 4, j = rem & 15;
    o[(((t * 4 + kk) * 32) + h2 * 16 + c) * 16 + j] = (bf16_t)w[idx];
  }
}

// ---------------------------------------------------------------------------
// in-degree via integer atomics
__global__ __launch_bounds__(256)
void degree_count(const int* __restrict__ dst, int* __restrict__ indeg, int nE) {
  int e = blockIdx.x * blockDim.x + threadIdx.x;
  if (e < nE) atomicAdd(&indeg[dst[e]], 1);
}

// ---------------------------------------------------------------------------
// h0 = value_proj_w[feat_id % FIXED] + value_proj_b + deg_emb[clip(indeg,0,1000)]
// also seeds agg = h0 (GIN self term).
__global__ __launch_bounds__(256)
void init_h(const int* __restrict__ feat, const int* __restrict__ indeg,
            const float* __restrict__ vpw, const float* __restrict__ vpb,
            const float* __restrict__ demb,
            float* __restrict__ h, float* __restrict__ agg) {
  int i = blockIdx.x * blockDim.x + threadIdx.x;
  if (i >= GN * 32) return;
  int v  = i >> 5;
  int c4 = (i & 31) << 2;
  int f = feat[v] % 10000; if (f < 0) f += 10000;
  int dg = indeg[v]; if (dg > 1000) dg = 1000;
  float4 w  = *(const float4*)&vpw[f * GD + c4];
  float4 b  = *(const float4*)&vpb[c4];
  float4 de = *(const float4*)&demb[dg * GD + c4];
  float4 o;
  o.x = w.x + b.x + de.x; o.y = w.y + b.y + de.y;
  o.z = w.z + b.z + de.z; o.w = w.w + b.w + de.w;
  *(float4*)&h[v * GD + c4]   = o;
  *(float4*)&agg[v * GD + c4] = o;
}

// ---------------------------------------------------------------------------
// agg[dst] += h[src]   (hardware global_atomic_add_f32; agg is L2-resident)
__global__ __launch_bounds__(256)
void scatter_add(const int* __restrict__ src, const int* __restrict__ dst,
                 const float* __restrict__ h, float* __restrict__ agg, int nE) {
  int i = blockIdx.x * blockDim.x + threadIdx.x;
  if (i >= nE * 32) return;
  int e  = i >> 5;
  int c4 = (i & 31) << 2;
  int s = src[e], d = dst[e];
  float4 v = *(const float4*)&h[s * GD + c4];
  float* p = &agg[d * GD + c4];
  __hip_atomic_fetch_add(p + 0, v.x, __ATOMIC_RELAXED, __HIP_MEMORY_SCOPE_AGENT);
  __hip_atomic_fetch_add(p + 1, v.y, __ATOMIC_RELAXED, __HIP_MEMORY_SCOPE_AGENT);
  __hip_atomic_fetch_add(p + 2, v.z, __ATOMIC_RELAXED, __HIP_MEMORY_SCOPE_AGENT);
  __hip_atomic_fetch_add(p + 3, v.w, __ATOMIC_RELAXED, __HIP_MEMORY_SCOPE_AGENT);
}

// ---------------------------------------------------------------------------
// Out[N,128] = act(A) @ W + bias, with fused per-channel sum / sumsq for BN.
// mode==1 applies the previous BN affine + ReLU while converting A to bf16.
// 128 threads = 4 wave32; each wave: 16 output rows, 8 col tiles, K=128 in
// 4 accumulator-chained v_wmma_f32_16x16x32_bf16 steps.
//   B fragments: direct from fragment-major packed global (2x global_load_b128)
//   A fragments: LDS rows in natural K order so fragment groups of 8 are both
//                contiguous in A (f32 b128 loads) and in LDS (b128 load/store)
__global__ __launch_bounds__(128)
void gin_gemm_bn(const float* __restrict__ A, const bf16_t* __restrict__ Wp,
                 const float* __restrict__ bias,
                 const float* __restrict__ scale, const float* __restrict__ shift,
                 float* __restrict__ Out, float* __restrict__ stats,
                 int nRows, int mode) {
  __shared__ bf16_t AsF[64][GD + 8];       // 17 KB, rows in natural K order (272B stride)
  __shared__ float  sSum[GD];
  __shared__ float  sSq[GD];

  const int tid     = threadIdx.x;
  const int lane    = tid & 31;
  const int wave    = tid >> 5;
  const int lr      = lane & 15;   // row (A) / col (B,D) within tile
  const int halfSel = lane >> 4;   // K half-group selector
  const int rowBase = blockIdx.x * 64;
  const bool fullTile = (rowBase + 64 <= nRows);   // block-uniform

  if (tid < GD) { sSum[tid] = 0.f; sSq[tid] = 0.f; }

  // A -> AsF: 8 fragment-groups of 8 contiguous channels per thread.
  // group q = kk*4 + h2*2 + u maps to channels Kb..Kb+7, Kb = kk*32+h2*16+u*8,
  // stored at AsF[r][q*8] so lane fragment loads are contiguous 32B.
  for (int g = tid; g < 64 * 16; g += 128) {
    int r = g >> 4, q = g & 15;
    int kk = q >> 2, h2 = (q >> 1) & 1, u = q & 1;
    int Kb = kk * 32 + h2 * 16 + u * 8;
    int m = rowBase + r;
    float4 x0, x1;
    if (fullTile || m < nRows) {
      x0 = *(const float4*)&A[m * GD + Kb];
      x1 = *(const float4*)&A[m * GD + Kb + 4];
    } else {
      x0.x = x0.y = x0.z = x0.w = 0.f;
      x1.x = x1.y = x1.z = x1.w = 0.f;
    }
    if (mode) {
      float4 s0 = *(const float4*)&scale[Kb];
      float4 s1 = *(const float4*)&scale[Kb + 4];
      float4 t0 = *(const float4*)&shift[Kb];
      float4 t1 = *(const float4*)&shift[Kb + 4];
      x0.x = fmaxf(s0.x * x0.x + t0.x, 0.f);
      x0.y = fmaxf(s0.y * x0.y + t0.y, 0.f);
      x0.z = fmaxf(s0.z * x0.z + t0.z, 0.f);
      x0.w = fmaxf(s0.w * x0.w + t0.w, 0.f);
      x1.x = fmaxf(s1.x * x1.x + t1.x, 0.f);
      x1.y = fmaxf(s1.y * x1.y + t1.y, 0.f);
      x1.z = fmaxf(s1.z * x1.z + t1.z, 0.f);
      x1.w = fmaxf(s1.w * x1.w + t1.w, 0.f);
    }
    v8bf o8;
    o8[0] = (bf16_t)x0.x; o8[1] = (bf16_t)x0.y;
    o8[2] = (bf16_t)x0.z; o8[3] = (bf16_t)x0.w;
    o8[4] = (bf16_t)x1.x; o8[5] = (bf16_t)x1.y;
    o8[6] = (bf16_t)x1.z; o8[7] = (bf16_t)x1.w;
    *(v8bf*)&AsF[r][q * 8] = o8;
  }
  __syncthreads();

  // A fragments for this wave's 16 rows: contiguous 32B per (kk) per lane.
  Frag32B afrag[4];
  {
    const bf16_t* ap = &AsF[wave * 16 + lr][halfSel * 16];
    #pragma unroll
    for (int kk = 0; kk < 4; ++kk) {
      afrag[kk].h[0] = *(const v8bf*)(ap + kk * 32);
      afrag[kk].h[1] = *(const v8bf*)(ap + kk * 32 + 8);
    }
  }

  const bf16_t* wl = Wp + lane * 16;   // per-lane base into packed fragments

  #pragma unroll
  for (int t = 0; t < 8; ++t) {
    v8f acc = {};
    #pragma unroll
    for (int kk = 0; kk < 4; ++kk) {
      Frag32B bfr;
      const bf16_t* bp = wl + (t * 4 + kk) * 512;   // 32 lanes * 16 elems
      bfr.h[0] = *(const v8bf*)bp;
      bfr.h[1] = *(const v8bf*)(bp + 8);
      acc = __builtin_amdgcn_wmma_f32_16x16x32_bf16(
          false, afrag[kk].v, false, bfr.v, (short)0, acc, false, false);
    }
    // D 16x16 f32: VGPR r -> M = r + 8*halfSel ; N = lr
    int ocol = t * 16 + lr;
    float bv = bias[ocol];
    float lsum = 0.f, lsq = 0.f;
    int m0 = rowBase + wave * 16 + halfSel * 8;
    if (fullTile) {                      // branch-free hot path (781/782 blocks)
      #pragma unroll
      for (int r = 0; r < 8; ++r) {
        float v = acc[r] + bv;
        Out[(m0 + r) * GD + ocol] = v;
        lsum += v; lsq += v * v;
      }
    } else {
      #pragma unroll
      for (int r = 0; r < 8; ++r) {
        if (m0 + r < nRows) {
          float v = acc[r] + bv;
          Out[(m0 + r) * GD + ocol] = v;
          lsum += v; lsq += v * v;
        }
      }
    }
    __hip_atomic_fetch_add(&sSum[ocol], lsum, __ATOMIC_RELAXED, __HIP_MEMORY_SCOPE_WORKGROUP);
    __hip_atomic_fetch_add(&sSq[ocol],  lsq,  __ATOMIC_RELAXED, __HIP_MEMORY_SCOPE_WORKGROUP);
  }
  __syncthreads();
  if (tid < GD) {
    __hip_atomic_fetch_add(&stats[tid],      sSum[tid], __ATOMIC_RELAXED, __HIP_MEMORY_SCOPE_AGENT);
    __hip_atomic_fetch_add(&stats[GD + tid], sSq[tid],  __ATOMIC_RELAXED, __HIP_MEMORY_SCOPE_AGENT);
  }
}

// ---------------------------------------------------------------------------
// stats -> per-channel affine (scale,shift) of training-mode BN; rezero stats.
__global__ __launch_bounds__(128)
void finalize_bn(float* __restrict__ stats, const float* __restrict__ g,
                 const float* __restrict__ b, float* __restrict__ scale,
                 float* __restrict__ shift, float invN) {
  int c = threadIdx.x;
  if (c < GD) {
    float mean = stats[c] * invN;
    float var  = stats[GD + c] * invN - mean * mean;
    var = var < 0.f ? 0.f : var;
    float inv = rsqrtf(var + 1e-5f);
    float s = g[c] * inv;
    scale[c] = s;
    shift[c] = b[c] - mean * s;
    stats[c] = 0.f;
    stats[GD + c] = 0.f;
  }
}

// ---------------------------------------------------------------------------
// h = relu(scale*y + shift); optionally seed agg = h for the next layer.
__global__ __launch_bounds__(256)
void bn_relu_h(const float* __restrict__ y, const float* __restrict__ scale,
               const float* __restrict__ shift, float* __restrict__ hOut,
               float* __restrict__ aggOut, int writeAgg) {
  int i = blockIdx.x * blockDim.x + threadIdx.x;
  if (i >= GN * 32) return;
  int c4   = (i & 31) << 2;
  int base = (i >> 5) * GD + c4;
  float4 x = *(const float4*)&y[base];
  float4 o;
  o.x = fmaxf(scale[c4 + 0] * x.x + shift[c4 + 0], 0.f);
  o.y = fmaxf(scale[c4 + 1] * x.y + shift[c4 + 1], 0.f);
  o.z = fmaxf(scale[c4 + 2] * x.z + shift[c4 + 2], 0.f);
  o.w = fmaxf(scale[c4 + 3] * x.w + shift[c4 + 3], 0.f);
  *(float4*)&hOut[base] = o;
  if (writeAgg) *(float4*)&aggOut[base] = o;
}

// ---------------------------------------------------------------------------
// per-graph sums + counts
__global__ __launch_bounds__(256)
void pool_sum(const float* __restrict__ h, const int* __restrict__ batch,
              float* __restrict__ gsum, int* __restrict__ gcnt) {
  int i = blockIdx.x * blockDim.x + threadIdx.x;
  if (i >= GN * 32) return;
  int v  = i >> 5;
  int c4 = (i & 31) << 2;
  int g = batch[v];
  float4 hv = *(const float4*)&h[v * GD + c4];
  float* p = &gsum[g * GD + c4];
  __hip_atomic_fetch_add(p + 0, hv.x, __ATOMIC_RELAXED, __HIP_MEMORY_SCOPE_AGENT);
  __hip_atomic_fetch_add(p + 1, hv.y, __ATOMIC_RELAXED, __HIP_MEMORY_SCOPE_AGENT);
  __hip_atomic_fetch_add(p + 2, hv.z, __ATOMIC_RELAXED, __HIP_MEMORY_SCOPE_AGENT);
  __hip_atomic_fetch_add(p + 3, hv.w, __ATOMIC_RELAXED, __HIP_MEMORY_SCOPE_AGENT);
  if ((i & 31) == 0) atomicAdd(&gcnt[g], 1);
}

__global__ __launch_bounds__(256)
void pool_finalize(const float* __restrict__ gsum, const int* __restrict__ gcnt,
                   float* __restrict__ out) {
  int i = blockIdx.x * blockDim.x + threadIdx.x;
  if (i >= GG * GD) return;
  int g = i >> 7;
  float c = (float)gcnt[g];
  out[i] = gsum[i] / (c > 1.f ? c : 1.f);
}

// ---------------------------------------------------------------------------
extern "C" void kernel_launch(void* const* d_in, const int* in_sizes, int n_in,
                              void* d_out, int out_size, void* d_ws, size_t ws_size,
                              hipStream_t stream) {
  (void)in_sizes; (void)n_in; (void)out_size; (void)ws_size;
  const int N = GN, E = GE, D = GD, L = GL, G = GG;

  const int*   feat  = (const int*)d_in[0];
  const int*   ei    = (const int*)d_in[1];
  const int*   src   = ei;
  const int*   dst   = ei + E;
  const int*   batch = (const int*)d_in[2];
  const float* vpw   = (const float*)d_in[3];
  const float* vpb   = (const float*)d_in[4];
  const float* demb  = (const float*)d_in[5];
  const float* w1    = (const float*)d_in[6];
  const float* b1    = (const float*)d_in[7];
  const float* g1    = (const float*)d_in[8];
  const float* bb1   = (const float*)d_in[9];
  const float* w2    = (const float*)d_in[10];
  const float* b2    = (const float*)d_in[11];
  const float* g2    = (const float*)d_in[12];
  const float* bb2   = (const float*)d_in[13];

  float* out    = (float*)d_out;
  float* hFinal = out + G * D;   // h goes straight into the output buffer

  // workspace layout (floats, then packed bf16 weights, then ints); ~77MB
  float* ws     = (float*)d_ws;
  size_t nd     = (size_t)N * D;
  float* hbuf   = ws;
  float* aggbuf = ws + nd;
  float* ybuf   = ws + 2 * nd;
  float* stats  = ws + 3 * nd;        // 256: sum[128], sumsq[128]
  float* scaleA = stats + 256;
  float* shiftA = scaleA + GD;
  float* scaleB = shiftA + GD;
  float* shiftB = scaleB + GD;
  float* gsum   = shiftB + GD;        // G*D
  bf16_t* wp1   = (bf16_t*)(gsum + G * D);           // L*D*D bf16
  bf16_t* wp2   = wp1 + (size_t)L * D * D;           // L*D*D bf16
  int*   indeg  = (int*)(wp2 + (size_t)L * D * D);
  int*   gcnt   = indeg + N;

  const int T = 256;
  zero_misc   <<<(N + T - 1) / T, T, 0, stream>>>(indeg, gsum, gcnt, stats);
  pack_w      <<<L, 128, 0, stream>>>(w1, wp1);      // fragment-major bf16 weights
  pack_w      <<<L, 128, 0, stream>>>(w2, wp2);
  degree_count<<<(E + T - 1) / T, T, 0, stream>>>(dst, indeg, E);
  init_h      <<<(N * 32 + T - 1) / T, T, 0, stream>>>(feat, indeg, vpw, vpb, demb,
                                                       hbuf, aggbuf);

  const float invN = 1.0f / (float)N;
  const int gemmBlocks = (N + 63) / 64;
  for (int i = 0; i < L; ++i) {
    scatter_add<<<(E * 32 + T - 1) / T, T, 0, stream>>>(src, dst, hbuf, aggbuf, E);
    // z1 = agg @ W1 + b1 (+BN1 stats)
    gin_gemm_bn<<<gemmBlocks, 128, 0, stream>>>(aggbuf, wp1 + i * D * D, b1 + i * D,
                                                scaleA, shiftA, ybuf, stats, N, 0);
    finalize_bn<<<1, 128, 0, stream>>>(stats, g1 + i * D, bb1 + i * D,
                                       scaleA, shiftA, invN);
    // z2 = relu(bn1(z1)) @ W2 + b2 (+BN2 stats); affine+relu fused into A load
    gin_gemm_bn<<<gemmBlocks, 128, 0, stream>>>(ybuf, wp2 + i * D * D, b2 + i * D,
                                                scaleA, shiftA, aggbuf, stats, N, 1);
    finalize_bn<<<1, 128, 0, stream>>>(stats, g2 + i * D, bb2 + i * D,
                                       scaleB, shiftB, invN);
    const int last = (i == L - 1);
    bn_relu_h<<<(N * 32 + T - 1) / T, T, 0, stream>>>(aggbuf, scaleB, shiftB,
                                                      last ? hFinal : hbuf,
                                                      aggbuf, last ? 0 : 1);
  }
  pool_sum     <<<(N * 32 + T - 1) / T, T, 0, stream>>>(hFinal, batch, gsum, gcnt);
  pool_finalize<<<(G * D + T - 1) / T, T, 0, stream>>>(gsum, gcnt, out);
}
